// SlidingWindowAttentionModule_89060441850296
// MI455X (gfx1250) — matrix operand order repaired and checked by reading, not compiled, not run on hardware
//
#include <hip/hip_runtime.h>

// ---------------------------------------------------------------------------
// Sliding-window attention for MI455X (gfx1250), bf16 WMMA pipeline.
//   x:[2,4096,2048] f32, w_qkv:[6144,2048] f32, w_out:[2048,2048] f32
//   out:[2,4096,2048] f32
// Stages: convert->bf16 | TDM-staged GEMM qkv | flash sliding-window attn
//         (TDM K staging) | TDM-staged GEMM out
// ---------------------------------------------------------------------------

typedef __attribute__((ext_vector_type(16))) __bf16 v16bf;
typedef __attribute__((ext_vector_type(8)))  __bf16 v8bf;
typedef __attribute__((ext_vector_type(8)))  float  v8f;
typedef __attribute__((ext_vector_type(4)))  unsigned int u32x4;
typedef __attribute__((ext_vector_type(8)))  int          i32x8;
typedef __attribute__((ext_vector_type(4)))  int          i32x4;

#define EMBED   2048
#define NHEADS  16
#define HD      128
#define WIN     512
#define BATCH   2
#define SEQ     4096
#define F3      (3 * EMBED)
#define NCHUNK  (SEQ / WIN)

__device__ __forceinline__ __bf16 f2bf(float f) {
    unsigned u = __builtin_bit_cast(unsigned, f);
    u += 0x7fffu + ((u >> 16) & 1u);               // round-to-nearest-even
    unsigned short h = (unsigned short)(u >> 16);
    return __builtin_bit_cast(__bf16, h);
}

__device__ __forceinline__ v8f zero8() {
    v8f z;
#pragma unroll
    for (int i = 0; i < 8; ++i) z[i] = 0.0f;
    return z;
}

// ---------------------------------------------------------------------------
// Tensor Data Mover: 2D bf16 tile load (global -> LDS) via D# descriptor.
// tile_d0 = contiguous elements per row, tile_d1 = rows, stride0 = row stride
// (elements). LDS rows padded: after every (2<<pad_int) dwords insert
// (pad_amt+1) dwords -> LDS row stride = tile_d0 + 2*(pad_amt+1) elements.
// tensor dims == tile dims (no OOB clipping needed; all shapes exact).
__device__ __forceinline__ void tdm_load_2d(const void* gaddr, unsigned lds_addr,
                                            unsigned tile_d0, unsigned tile_d1,
                                            unsigned long long stride0_elems,
                                            unsigned pad_int, unsigned pad_amt) {
    unsigned long long ga = (unsigned long long)(uintptr_t)gaddr;
    u32x4 g0;
    g0[0] = 1u;                                          // count=1, user desc
    g0[1] = lds_addr;                                    // LDS byte address
    g0[2] = (unsigned)(ga & 0xffffffffu);                // global_addr[31:0]
    g0[3] = (unsigned)((ga >> 32) & 0x1ffffffu)          // global_addr[56:32]
          | (2u << 30);                                  // type=2 ("image")
    i32x8 g1;
    g1[0] = (int)((1u << 16)                             // data_size=1 (2B)
                | (1u << 20)                             // pad_enable
                | (pad_int << 22)                        // pad_interval
                | (pad_amt << 25));                      // pad_amount
    g1[1] = (int)((tile_d0 & 0xffffu) << 16);            // tensor_dim0[15:0]
    g1[2] = (int)((tile_d0 >> 16) | ((tile_d1 & 0xffffu) << 16));
    g1[3] = (int)((tile_d1 >> 16) | (tile_d0 << 16));    // tile_dim0
    g1[4] = (int)(tile_d1 & 0xffffu);                    // tile_dim1 (dim2=0)
    g1[5] = (int)(stride0_elems & 0xffffffffu);          // dim0_stride[31:0]
    g1[6] = (int)((stride0_elems >> 32) & 0xffffu);      // dim0_stride[47:32]
    g1[7] = 0;
    i32x4 z4 = {0, 0, 0, 0};
    i32x8 z8 = {0, 0, 0, 0, 0, 0, 0, 0};
    __builtin_amdgcn_tensor_load_to_lds(g0, g1, z4, z4, z8, 0);
}

// ---------------------------------------------------------------------------
// A fragment (16x32 bf16, M x K). Source row-major [rows, ld], contraction
// axis contiguous. Per ISA 16-bit A layout (VGPR0: K0,1 | K8,9 etc).
__device__ __forceinline__ v16bf load_frag_a(const __bf16* src, int ld, int k0, int lane) {
    int half = lane >> 4;
    int m    = lane & 15;
    const __bf16* p = src + (size_t)m * ld + k0 + half * 8;
    v8bf lo = *(const v8bf*)p;
    v8bf hi = *(const v8bf*)(p + 16);
    v16bf r;
#pragma unroll
    for (int i = 0; i < 8; ++i) { r[i] = lo[i]; r[i + 8] = hi[i]; }
    return r;
}

// B fragment (32x16 bf16, K x N). Source laid out [N, K] row-major.
// Lanes 0-15: N=l, K=0..7,16..23; lanes 16-31: N=l-16, K=8..15,24..31.
__device__ __forceinline__ v16bf load_frag_b(const __bf16* src, int ld, int k0, int lane) {
    int half = lane >> 4;
    int n    = lane & 15;
    const __bf16* p = src + (size_t)n * ld + k0 + half * 8;
    v8bf lo = *(const v8bf*)p;
    v8bf hi = *(const v8bf*)(p + 16);
    v16bf r;
#pragma unroll
    for (int i = 0; i < 8; ++i) { r[i] = lo[i]; r[i + 8] = hi[i]; }
    return r;
}

__device__ __forceinline__ v8f wmma_bf16(v16bf a, v16bf b, v8f c) {
    return __builtin_amdgcn_wmma_f32_16x16x32_bf16(false, a, false, b, (short)0, c,
                                                   false, false);
}

// ---------------------------------------------------------------------------
__global__ void convert_f32_bf16(const float* __restrict__ in,
                                 __bf16* __restrict__ out, size_t n) {
    size_t i = (size_t)blockIdx.x * blockDim.x + threadIdx.x;
    if (i < n) out[i] = f2bf(in[i]);
}

// ---------------------------------------------------------------------------
// C[m,n] = sum_k A[m,k] * B[n,k]  (both row-major, contraction innermost).
// Block tile 128x128, K stepped 32. 8 waves arranged 2(M)x4(N): wave tile
// 64x32 = 4x2 WMMA tiles. A/B tiles DMA'd into double-buffered LDS by the
// Tensor Data Mover (wave 0 issues, TENSORcnt + barrier synchronizes), rows
// hardware-padded to stride 40 elements (16 dwords + 4 pad dwords).
template <bool BF16OUT>
__global__ __launch_bounds__(256) void gemm_nt(const __bf16* __restrict__ A,
                                               const __bf16* __restrict__ B,
                                               float* __restrict__ Cf,
                                               __bf16* __restrict__ Cb,
                                               int M, int N, int K) {
    __shared__ __align__(16) __bf16 Ash[2][128][40];
    __shared__ __align__(16) __bf16 Bsh[2][128][40];

    const int lane = threadIdx.x & 31;
    const int wave = threadIdx.x >> 5;
    const int wrow = wave >> 2;               // 0..1 -> 64 rows each
    const int wcol = wave & 3;                // 0..3 -> 32 cols each
    const int row0 = blockIdx.y * 128;
    const int col0 = blockIdx.x * 128;

    const __bf16* Ab = A + (size_t)row0 * K;
    const __bf16* Bb = B + (size_t)col0 * K;

    if (wave == 0) {
        tdm_load_2d(Ab, (unsigned)(uintptr_t)&Ash[0][0][0], 32, 128,
                    (unsigned long long)K, 3, 3);
        tdm_load_2d(Bb, (unsigned)(uintptr_t)&Bsh[0][0][0], 32, 128,
                    (unsigned long long)K, 3, 3);
        __builtin_amdgcn_s_wait_tensorcnt(0);
    }
    __syncthreads();

    v8f acc[4][2];
#pragma unroll
    for (int mi = 0; mi < 4; ++mi)
#pragma unroll
        for (int ni = 0; ni < 2; ++ni) acc[mi][ni] = zero8();

    int buf = 0;
    for (int k0 = 0; k0 < K; k0 += 32) {
        int nb = buf ^ 1;
        if (k0 + 32 < K && wave == 0) {       // prefetch next K-slab via TDM
            tdm_load_2d(Ab + k0 + 32, (unsigned)(uintptr_t)&Ash[nb][0][0],
                        32, 128, (unsigned long long)K, 3, 3);
            tdm_load_2d(Bb + k0 + 32, (unsigned)(uintptr_t)&Bsh[nb][0][0],
                        32, 128, (unsigned long long)K, 3, 3);
        }

        v16bf afr[4], bfr[2];
#pragma unroll
        for (int mi = 0; mi < 4; ++mi)
            afr[mi] = load_frag_a(&Ash[buf][wrow * 64 + mi * 16][0], 40, 0, lane);
#pragma unroll
        for (int ni = 0; ni < 2; ++ni)
            bfr[ni] = load_frag_b(&Bsh[buf][wcol * 32 + ni * 16][0], 40, 0, lane);
#pragma unroll
        for (int mi = 0; mi < 4; ++mi)
#pragma unroll
            for (int ni = 0; ni < 2; ++ni)
                acc[mi][ni] = wmma_bf16(afr[mi], bfr[ni], acc[mi][ni]);

        if (wave == 0) __builtin_amdgcn_s_wait_tensorcnt(0);
        __syncthreads();
        buf = nb;
    }

    const int half = lane >> 4;
    const int nn   = lane & 15;
#pragma unroll
    for (int mi = 0; mi < 4; ++mi) {
#pragma unroll
        for (int ni = 0; ni < 2; ++ni) {
#pragma unroll
            for (int r = 0; r < 8; ++r) {
                int row = row0 + wrow * 64 + mi * 16 + r + 8 * half;
                int col = col0 + wcol * 32 + ni * 16 + nn;
                size_t idx = (size_t)row * N + col;
                if (BF16OUT) Cb[idx] = f2bf(acc[mi][ni][r]);
                else         Cf[idx] = acc[mi][ni][r];
            }
        }
    }
}

// ---------------------------------------------------------------------------
// Flash sliding-window attention over the qkv bf16 buffer [B,S,3E].
// Grid block = (b, h, chunk, qb): 8 waves x 16 query rows = 128 rows/block.
// kv span-local index j in [0, 1024); valid iff j>=i && j<=i+WIN (chunk 0
// starts at j=512 so global kv rows are always >= 0). K tile staged by TDM
// (rows padded to 136), V tile transposed manually into LDS.
__global__ __launch_bounds__(256) void swattn_kernel(const __bf16* __restrict__ qkv,
                                                     __bf16* __restrict__ attn) {
    __shared__ __align__(16) __bf16 Ksh[32][136];     // [j][d], TDM-padded
    __shared__ __align__(16) __bf16 Vt[128][40];      // [d][j] transposed
    __shared__ __align__(16) __bf16 Psh[8][16][40];   // per-wave P patch

    const int tid  = threadIdx.x;
    const int lane = tid & 31;
    const int wave = tid >> 5;
    const int half = lane >> 4;
    const int nn   = lane & 15;

    int idx   = blockIdx.x;
    int qb    = idx & 3;
    int chunk = (idx >> 2) & 7;
    int h     = (idx >> 5) & 15;
    int b     = idx >> 9;

    const int qlo = qb * 128;          // chunk-local base of this block's rows
    const int iw  = qlo + wave * 16;   // chunk-local base of this wave's rows
    const float scale = 0.08838834764831845f;  // 1/sqrt(128)

    // Preload Q fragments (16 rows x 128 d) as 4 A-fragments, reused all steps.
    const __bf16* qbase =
        qkv + ((size_t)b * SEQ + (size_t)chunk * WIN + iw) * F3 + (size_t)h * HD;
    v16bf qa[4];
#pragma unroll
    for (int t = 0; t < 4; ++t) qa[t] = load_frag_a(qbase, F3, t * 32, lane);

    v8f o[8];
#pragma unroll
    for (int t = 0; t < 8; ++t) o[t] = zero8();
    float mrun[8], lrun[8];
#pragma unroll
    for (int r = 0; r < 8; ++r) { mrun[r] = -INFINITY; lrun[r] = 0.0f; }

    const int jstart = (chunk == 0) ? 512 : qlo;
    const int jend   = qlo + 640;     // exclusive: covers i+WIN for all rows

    for (int j0 = jstart; j0 < jend; j0 += 32) {
        const int grow = (chunk - 1) * WIN + j0;       // always in [0, SEQ)
        // --- K tile [32][128] via Tensor Data Mover -------------------------
        if (wave == 0) {
            const __bf16* kbase =
                qkv + ((size_t)b * SEQ + grow) * F3 + EMBED + (size_t)h * HD;
            tdm_load_2d(kbase, (unsigned)(uintptr_t)&Ksh[0][0], 128, 32,
                        (unsigned long long)F3, 5, 3);   // pad 64+4 dw -> 136
        }
        // --- V tile transposed (manual; TDM has no transpose path) ----------
        {
            int jj = tid >> 3;                 // 0..31
            int c0 = (tid & 7) * 16;           // 0..112
            const __bf16* vrow =
                qkv + ((size_t)b * SEQ + grow + jj) * F3 + 2 * EMBED +
                (size_t)h * HD + c0;
            v8bf v0 = *(const v8bf*)vrow;
            v8bf v1 = *(const v8bf*)(vrow + 8);
#pragma unroll
            for (int i = 0; i < 8; ++i) {
                Vt[c0 + i][jj]     = v0[i];
                Vt[c0 + 8 + i][jj] = v1[i];
            }
        }
        if (wave == 0) __builtin_amdgcn_s_wait_tensorcnt(0);
        __syncthreads();

        // wave-uniform band test
        bool active = (j0 + 31 >= iw) && (j0 <= iw + 15 + WIN);
        if (active) {
            // ---- S = Q K^T (two 16x16 j-tiles, contraction over d=128) -----
            v8f s0 = zero8(), s1 = zero8();
#pragma unroll
            for (int t = 0; t < 4; ++t) {
                v16bf kb0 = load_frag_b(&Ksh[0][0], 136, t * 32, lane);
                v16bf kb1 = load_frag_b(&Ksh[16][0], 136, t * 32, lane);
                s0 = wmma_bf16(qa[t], kb0, s0);
                s1 = wmma_bf16(qa[t], kb1, s1);
            }

            // ---- masked online softmax (rows live in 16-lane halves) -------
            float p0[8], p1[8];
#pragma unroll
            for (int r = 0; r < 8; ++r) {
                int i   = iw + r + 8 * half;    // chunk-local query row
                int jc0 = j0 + nn;
                int jc1 = j0 + 16 + nn;
                float sv0 = s0[r] * scale;
                float sv1 = s1[r] * scale;
                if (!(jc0 >= i && jc0 <= i + WIN)) sv0 = -INFINITY;
                if (!(jc1 >= i && jc1 <= i + WIN)) sv1 = -INFINITY;

                float rm = fmaxf(sv0, sv1);
                rm = fmaxf(rm, __shfl_xor(rm, 1, 32));
                rm = fmaxf(rm, __shfl_xor(rm, 2, 32));
                rm = fmaxf(rm, __shfl_xor(rm, 4, 32));
                rm = fmaxf(rm, __shfl_xor(rm, 8, 32));

                float mold = mrun[r];
                float mnew = fmaxf(mold, rm);
                float alpha, e0, e1;
                if (mnew <= -1e30f) {          // nothing valid yet this row
                    alpha = 1.0f; e0 = 0.0f; e1 = 0.0f;
                } else {
                    alpha = (mold <= -1e30f) ? 0.0f : __expf(mold - mnew);
                    e0 = (sv0 <= -1e30f) ? 0.0f : __expf(sv0 - mnew);
                    e1 = (sv1 <= -1e30f) ? 0.0f : __expf(sv1 - mnew);
                }
                float rs = e0 + e1;
                rs += __shfl_xor(rs, 1, 32);
                rs += __shfl_xor(rs, 2, 32);
                rs += __shfl_xor(rs, 4, 32);
                rs += __shfl_xor(rs, 8, 32);

                mrun[r] = mnew;
                lrun[r] = lrun[r] * alpha + rs;
#pragma unroll
                for (int t = 0; t < 8; ++t) o[t][r] *= alpha;
                p0[r] = e0;
                p1[r] = e1;
            }

            // ---- round P to bf16 via wave-private LDS patch ----------------
#pragma unroll
            for (int r = 0; r < 8; ++r) {
                int m = r + 8 * half;
                Psh[wave][m][nn]      = f2bf(p0[r]);
                Psh[wave][m][16 + nn] = f2bf(p1[r]);
            }
            asm volatile("s_wait_dscnt 0" ::: "memory");  // DS RAW, same wave

            // ---- O += P V (A = P 16x32, B = Vt[d][j] as [N,K]) -------------
            v16bf pa = load_frag_a(&Psh[wave][0][0], 40, 0, lane);
#pragma unroll
            for (int t = 0; t < 8; ++t) {
                v16bf vb = load_frag_b(&Vt[t * 16][0], 40, 0, lane);
                o[t] = wmma_bf16(pa, vb, o[t]);
            }
        }
        __syncthreads();
    }

    // ---- normalize and write attn[b, s, h*128 + d] as bf16 -----------------
    float inv[8];
#pragma unroll
    for (int r = 0; r < 8; ++r) inv[r] = 1.0f / lrun[r];
#pragma unroll
    for (int t = 0; t < 8; ++t) {
#pragma unroll
        for (int r = 0; r < 8; ++r) {
            int is = chunk * WIN + iw + r + 8 * half;   // chunk-local -> s
            size_t dst = ((size_t)b * SEQ + is) * EMBED +
                         (size_t)h * HD + t * 16 + nn;
            attn[dst] = f2bf(o[t][r] * inv[r]);
        }
    }
}

// ---------------------------------------------------------------------------
extern "C" void kernel_launch(void* const* d_in, const int* in_sizes, int n_in,
                              void* d_out, int out_size, void* d_ws, size_t ws_size,
                              hipStream_t stream) {
    const float* x    = (const float*)d_in[0];
    const float* wqkv = (const float*)d_in[1];
    const float* wout = (const float*)d_in[2];
    float* out = (float*)d_out;

    const size_t M    = (size_t)BATCH * SEQ;      // 8192
    const size_t nX   = M * EMBED;                // 16,777,216
    const size_t nWq  = (size_t)F3 * EMBED;       // 12,582,912
    const size_t nWo  = (size_t)EMBED * EMBED;    //  4,194,304
    const size_t nQKV = M * F3;                   // 50,331,648
    const size_t nA   = M * EMBED;

    char* ws = (char*)d_ws;
    size_t off = 0;
    __bf16* Xb   = (__bf16*)(ws + off); off += nX   * 2;
    __bf16* Wqb  = (__bf16*)(ws + off); off += nWq  * 2;
    __bf16* Wob  = (__bf16*)(ws + off); off += nWo  * 2;
    __bf16* QKVb = (__bf16*)(ws + off); off += nQKV * 2;
    __bf16* Ab   = (__bf16*)(ws + off); off += nA   * 2;
    if (ws_size < off) return;   // ~192 MB required

    convert_f32_bf16<<<(unsigned)((nX  + 255) / 256), 256, 0, stream>>>(x,    Xb,  nX);
    convert_f32_bf16<<<(unsigned)((nWq + 255) / 256), 256, 0, stream>>>(wqkv, Wqb, nWq);
    convert_f32_bf16<<<(unsigned)((nWo + 255) / 256), 256, 0, stream>>>(wout, Wob, nWo);

    // qkv[b,s,f] = sum_e x[b,s,e] w_qkv[f,e]    (M=8192, N=6144, K=2048)
    gemm_nt<true><<<dim3(F3 / 128, (unsigned)(M / 128)), 256, 0, stream>>>(
        Xb, Wqb, nullptr, QKVb, (int)M, F3, EMBED);

    // sliding-window attention -> Ab[b,s,e]
    swattn_kernel<<<BATCH * NHEADS * NCHUNK * 4, 256, 0, stream>>>(QKVb, Ab);

    // out[b,s,f] = sum_e attn[b,s,e] w_out[f,e] (M=8192, N=2048, K=2048)
    gemm_nt<false><<<dim3(EMBED / 128, (unsigned)(M / 128)), 256, 0, stream>>>(
        Ab, Wob, out, nullptr, (int)M, EMBED, EMBED);
}
